// AnomalyAttention_1752346657396
// MI455X (gfx1250) — compile-verified, matrix operand order checked by reading, not run on hardware
//
#include <hip/hip_runtime.h>
#include <hip/hip_bf16.h>
#include <math.h>

#define NTOK 4096
#define DMOD 512

typedef __attribute__((ext_vector_type(16))) _Float16 v16h;
typedef __attribute__((ext_vector_type(8)))  _Float16 v8h;
typedef __attribute__((ext_vector_type(4)))  _Float16 v4h;
typedef __attribute__((ext_vector_type(8)))  float    v8f;
typedef __attribute__((ext_vector_type(4)))  float    v4f;
typedef int vi4 __attribute__((vector_size(16)));   // int4, matches builtin sig

// ---------------------------------------------------------------------------
// gfx1250 async global->LDS path (ASYNCcnt), with compile-safe fallback.
// Builtin signature (from hipcc diagnostic): arg0 = int4* in AS(1) (global
// source), arg1 = LDS dest, then imm offset + imm cpol.
// ---------------------------------------------------------------------------
#if defined(__has_builtin)
#if __has_builtin(__builtin_amdgcn_global_load_async_to_lds_b128)
#define ASYNC_LDS 1
#endif
#endif

#if defined(ASYNC_LDS)
typedef __attribute__((address_space(1))) vi4* as1_vi4_ptr;
typedef __attribute__((address_space(3))) vi4* as3_vi4_ptr;
#if __has_builtin(__builtin_amdgcn_s_wait_asynccnt)
#define WAIT_ASYNC(n) __builtin_amdgcn_s_wait_asynccnt(n)
#else
#define WAIT_ASYNC(n) asm volatile("s_wait_asynccnt %0" ::"i"(n) : "memory")
#endif
#else
#define WAIT_ASYNC(n)
#endif

// ---------------------------------------------------------------------------
// Fragment loaders.
// CDNA5 16-bit A/B fragment (16x32 MxK / 32x16 KxN, NT form): per lane,
// elements 0..7  = K (grp?8:0)+0..7,  elements 8..15 = K 16+(grp?8:0)+0..7,
// where grp = lane>>4 and row/col index = lane&15.  Both chunks are
// contiguous 16-byte runs -> two b128 loads (global or ds).
// ---------------------------------------------------------------------------
__device__ __forceinline__ v16h load_frag_f16(const _Float16* p) {
    union { v16h v; v8h h[2]; } u;
    u.h[0] = *(const v8h*)(p);
    u.h[1] = *(const v8h*)(p + 16);
    return u.v;
}

// Same fragment but source data is fp32 (normalized S): 4x float4 loads,
// convert to f16 in-register.
__device__ __forceinline__ v16h load_frag_f32cvt(const float* p) {
    v4f f0 = *(const v4f*)(p);
    v4f f1 = *(const v4f*)(p + 4);
    v4f f2 = *(const v4f*)(p + 16);
    v4f f3 = *(const v4f*)(p + 20);
    v16h r;
#pragma unroll
    for (int e = 0; e < 4; ++e) {
        r[e]      = (_Float16)f0[e];
        r[4 + e]  = (_Float16)f1[e];
        r[8 + e]  = (_Float16)f2[e];
        r[12 + e] = (_Float16)f3[e];
    }
    return r;
}

// ---------------------------------------------------------------------------
// Stage a 64-row x 32-half B strip into LDS.  128 threads, each moves two
// 16-byte chunks.  Async path: per-lane global_load_async_to_lds_b128
// (ASYNCcnt); fallback: global load -> ds_store.
// ---------------------------------------------------------------------------
__device__ __forceinline__ void stage_b(const _Float16* __restrict__ g, int ldg,
                                        _Float16* l, int tid) {
#pragma unroll
    for (int c = tid; c < 256; c += 128) {
        int row  = c >> 2;
        int part = c & 3;
        const _Float16* gp = g + (size_t)row * ldg + part * 8;
        _Float16*       lp = l + row * 32 + part * 8;
#if defined(ASYNC_LDS)
        __builtin_amdgcn_global_load_async_to_lds_b128(
            (as1_vi4_ptr)(_Float16*)gp, (as3_vi4_ptr)lp, 0, 0);
#else
        *(v8h*)lp = *(const v8h*)gp;
#endif
    }
}

// ---------------------------------------------------------------------------
// K0: fp32 -> f16 conversion (vectorized x4)
// ---------------------------------------------------------------------------
__global__ void cvt_f32_to_f16_x4(const float* __restrict__ in,
                                  _Float16* __restrict__ out, int n4) {
    int idx = blockIdx.x * blockDim.x + threadIdx.x;
    if (idx < n4) {
        v4f f = ((const v4f*)in)[idx];
        v4h h;
#pragma unroll
        for (int e = 0; e < 4; ++e) h[e] = (_Float16)f[e];
        ((v4h*)out)[idx] = h;
    }
}

// ---------------------------------------------------------------------------
// sigma[i] = clip(x[i,:] . Ws, 0.001, 1.0)   (exact fp32, one wave per row)
// ---------------------------------------------------------------------------
__global__ void sigma_kernel(const float* __restrict__ x,
                             const float* __restrict__ Ws,
                             float* __restrict__ sigma) {
    int row  = (blockIdx.x * blockDim.x + threadIdx.x) >> 5;
    int lane = threadIdx.x & 31;
    const float* xr = x + (size_t)row * DMOD;
    float s = 0.f;
#pragma unroll 4
    for (int k = lane; k < DMOD; k += 32) s += xr[k] * Ws[k];
#pragma unroll
    for (int m = 16; m >= 1; m >>= 1) s += __shfl_xor(s, m, 32);
    if (lane == 0) sigma[row] = fminf(fmaxf(s, 0.001f), 1.0f);
}

// ---------------------------------------------------------------------------
// K1: QKV projection.  Q[i,c] = sum_k x[i,k] * W[c,k]  (NT GEMM, f16->f32).
// blockIdx.z selects Wq/Wk/Wv.  Q,K stored row-major f16; V stored TRANSPOSED
// (Vt[c*N + i]) so that S@V later is also an NT GEMM with b128 loads.
// W is only 512 KB and L2-hot, so direct global loads are fine here.
// ---------------------------------------------------------------------------
__global__ void qkv_kernel(const _Float16* __restrict__ xh,
                           const _Float16* __restrict__ wh_all,
                           _Float16* __restrict__ qh,
                           _Float16* __restrict__ kmh,
                           _Float16* __restrict__ vt) {
    const int i0    = blockIdx.x * 16;
    const int c0    = blockIdx.y * 64;
    const int which = blockIdx.z;
    const _Float16* wh = wh_all + (size_t)which * DMOD * DMOD;

    const int lane = threadIdx.x;
    const int grp  = lane >> 4;
    const int ln   = lane & 15;

    v8f acc[4] = {v8f{}, v8f{}, v8f{}, v8f{}};
    const _Float16* arow = xh + (size_t)(i0 + ln) * DMOD + grp * 8;

    for (int k = 0; k < DMOD; k += 32) {
        v16h a = load_frag_f16(arow + k);
#pragma unroll
        for (int t = 0; t < 4; ++t) {
            v16h b = load_frag_f16(wh + (size_t)(c0 + t * 16 + ln) * DMOD + grp * 8 + k);
            acc[t] = __builtin_amdgcn_wmma_f32_16x16x32_f16(
                false, a, false, b, (short)0, acc[t], false, false);
        }
    }

    if (which < 2) {
        _Float16* out = (which == 0) ? qh : kmh;
#pragma unroll
        for (int t = 0; t < 4; ++t)
#pragma unroll
            for (int r = 0; r < 8; ++r)
                out[(size_t)(i0 + grp * 8 + r) * DMOD + c0 + t * 16 + ln] =
                    (_Float16)acc[t][r];
    } else {
#pragma unroll
        for (int t = 0; t < 4; ++t) {
            v8h pk;
#pragma unroll
            for (int r = 0; r < 8; ++r) pk[r] = (_Float16)acc[t][r];
            *(v8h*)(vt + (size_t)(c0 + t * 16 + ln) * NTOK + i0 + grp * 8) = pk;
        }
    }
}

// ---------------------------------------------------------------------------
// K2a: E = exp(Q.K^T / sqrt(d)) into S region + per-row sums (f32 atomics).
// 4-wave workgroup covers a 64x64 tile; the shared 64-row K strip is
// double-buffered in LDS via async global->LDS copies.  Scores ~N(0,1):
// no max-subtraction needed in fp32.
// ---------------------------------------------------------------------------
__global__ void scores_kernel(const _Float16* __restrict__ qh,
                              const _Float16* __restrict__ kmh,
                              float* __restrict__ S,
                              float* __restrict__ rowsum) {
    __shared__ _Float16 sB[2][64 * 32];

    const int tid  = threadIdx.x;
    const int wv   = tid >> 5;
    const int lane = tid & 31;
    const int grp  = lane >> 4, ln = lane & 15;

    const int i0 = blockIdx.x * 64 + wv * 16;   // this wave's rows
    const int j0 = blockIdx.y * 64;             // block's cols

    v8f acc[4] = {v8f{}, v8f{}, v8f{}, v8f{}};
    const _Float16* arow  = qh + (size_t)(i0 + ln) * DMOD + grp * 8;
    const _Float16* bbase = kmh + (size_t)j0 * DMOD;

    stage_b(bbase, DMOD, &sB[0][0], tid);
    for (int kk = 0; kk < DMOD; kk += 32) {
        const int  buf  = (kk >> 5) & 1;
        const bool more = (kk + 32) < DMOD;
        if (more) stage_b(bbase + kk + 32, DMOD, &sB[buf ^ 1][0], tid);
        if (more) { WAIT_ASYNC(2); } else { WAIT_ASYNC(0); }
        __syncthreads();

        v16h a = load_frag_f16(arow + kk);
#pragma unroll
        for (int t = 0; t < 4; ++t) {
            v16h b = load_frag_f16(&sB[buf][(t * 16 + ln) * 32 + grp * 8]);
            acc[t] = __builtin_amdgcn_wmma_f32_16x16x32_f16(
                false, a, false, b, (short)0, acc[t], false, false);
        }
        __syncthreads();   // buffer free before it is re-staged
    }

    const float scale = 0.04419417382415922f; // 1/sqrt(512)
    float rs[8] = {0, 0, 0, 0, 0, 0, 0, 0};
#pragma unroll
    for (int t = 0; t < 4; ++t)
#pragma unroll
        for (int r = 0; r < 8; ++r) {
            float e = __expf(acc[t][r] * scale);
            S[(size_t)(i0 + grp * 8 + r) * NTOK + j0 + t * 16 + ln] = e;
            rs[r] += e;
        }
#pragma unroll
    for (int r = 0; r < 8; ++r) {
        float v = rs[r];
#pragma unroll
        for (int m = 1; m < 16; m <<= 1) v += __shfl_xor(v, m, 32);
        if (ln == 0) atomicAdd(&rowsum[i0 + grp * 8 + r], v);
    }
}

// ---------------------------------------------------------------------------
// K2b: S /= rowsum   (float4-vectorized, in place)
// ---------------------------------------------------------------------------
__global__ void snorm_kernel(float* __restrict__ S,
                             const float* __restrict__ rowsum) {
    size_t idx = (size_t)blockIdx.x * blockDim.x + threadIdx.x; // float4 units
    int row = (int)(idx >> 10);                                 // 1024 quads/row
    float inv = 1.0f / rowsum[row];
    v4f v = ((v4f*)S)[idx];
    v *= inv;
    ((v4f*)S)[idx] = v;
}

// ---------------------------------------------------------------------------
// K3: Z = S @ V.  Same 64x64-per-block structure; Vt strip async-staged into
// LDS; A-frag = fp32 S converted in-register.  fp32 output.
// ---------------------------------------------------------------------------
__global__ void zout_kernel(const float* __restrict__ S,
                            const _Float16* __restrict__ vt,
                            float* __restrict__ Z) {
    __shared__ _Float16 sB[2][64 * 32];

    const int tid  = threadIdx.x;
    const int wv   = tid >> 5;
    const int lane = tid & 31;
    const int grp  = lane >> 4, ln = lane & 15;

    const int i0 = blockIdx.x * 64 + wv * 16;
    const int c0 = blockIdx.y * 64;

    v8f acc[4] = {v8f{}, v8f{}, v8f{}, v8f{}};
    const float*    srow  = S + (size_t)(i0 + ln) * NTOK + grp * 8;
    const _Float16* bbase = vt + (size_t)c0 * NTOK;

    stage_b(bbase, NTOK, &sB[0][0], tid);
    for (int j = 0; j < NTOK; j += 32) {
        const int  buf  = (j >> 5) & 1;
        const bool more = (j + 32) < NTOK;
        if (more) stage_b(bbase + j + 32, NTOK, &sB[buf ^ 1][0], tid);
        if (more) { WAIT_ASYNC(2); } else { WAIT_ASYNC(0); }
        __syncthreads();

        v16h a = load_frag_f32cvt(srow + j);
#pragma unroll
        for (int t = 0; t < 4; ++t) {
            v16h b = load_frag_f16(&sB[buf][(t * 16 + ln) * 32 + grp * 8]);
            acc[t] = __builtin_amdgcn_wmma_f32_16x16x32_f16(
                false, a, false, b, (short)0, acc[t], false, false);
        }
        __syncthreads();
    }
#pragma unroll
    for (int t = 0; t < 4; ++t)
#pragma unroll
        for (int r = 0; r < 8; ++r)
            Z[(size_t)(i0 + grp * 8 + r) * DMOD + c0 + t * 16 + ln] = acc[t][r];
}

// ---------------------------------------------------------------------------
// K4: prior association P.  One 256-thread block per row; exp in registers,
// LDS tree reduction for the row denominator.
// ---------------------------------------------------------------------------
__global__ void prior_kernel(const float* __restrict__ sigma,
                             float* __restrict__ P) {
    const int i   = blockIdx.x;
    const int tid = threadIdx.x;
    const float sg   = sigma[i];
    const float coef = rsqrtf(6.283185307179586f * sg);
    const float aq   = -0.5f / (sg * sg);

    float g[16];
    float s = 0.f;
#pragma unroll
    for (int t = 0; t < 16; ++t) {
        int j = tid + t * 256;
        float d = fabsf((float)(i - j));
        float e = coef * __expf(aq * d * d);
        g[t] = e;
        s += e;
    }
    __shared__ float red[256];
    red[tid] = s;
    __syncthreads();
    for (int off = 128; off > 0; off >>= 1) {
        if (tid < off) red[tid] += red[tid + off];
        __syncthreads();
    }
    const float invd = 1.0f / (red[0] + 1e-8f);
#pragma unroll
    for (int t = 0; t < 16; ++t)
        P[(size_t)i * NTOK + tid + t * 256] = g[t] * invd;
}

// ---------------------------------------------------------------------------
// Host-side orchestration
// ---------------------------------------------------------------------------
extern "C" void kernel_launch(void* const* d_in, const int* in_sizes, int n_in,
                              void* d_out, int out_size, void* d_ws, size_t ws_size,
                              hipStream_t stream) {
    const float* x  = (const float*)d_in[0];
    const float* Wq = (const float*)d_in[1];
    const float* Wk = (const float*)d_in[2];
    const float* Wv = (const float*)d_in[3];
    const float* Ws = (const float*)d_in[4];

    // output layout: Z [N,D] | P [N,N] | S [N,N]
    float* Z = (float*)d_out;
    float* P = Z + (size_t)NTOK * DMOD;
    float* S = P + (size_t)NTOK * NTOK;

    // workspace layout
    char* ws = (char*)d_ws;
    _Float16* xh  = (_Float16*)ws;                         ws += (size_t)NTOK * DMOD * 2;
    _Float16* wh  = (_Float16*)ws;                         ws += (size_t)3 * DMOD * DMOD * 2;
    _Float16* qh  = (_Float16*)ws;                         ws += (size_t)NTOK * DMOD * 2;
    _Float16* kmh = (_Float16*)ws;                         ws += (size_t)NTOK * DMOD * 2;
    _Float16* vt  = (_Float16*)ws;                         ws += (size_t)NTOK * DMOD * 2;   // [D,N]
    float*    sigma  = (float*)ws;                         ws += (size_t)NTOK * 4;
    float*    rowsum = (float*)ws;

    // K0: fp32 -> f16 conversions
    {
        int n4 = NTOK * DMOD / 4;
        cvt_f32_to_f16_x4<<<n4 / 256, 256, 0, stream>>>(x, xh, n4);
        int w4 = DMOD * DMOD / 4;
        cvt_f32_to_f16_x4<<<w4 / 256, 256, 0, stream>>>(Wq, wh, w4);
        cvt_f32_to_f16_x4<<<w4 / 256, 256, 0, stream>>>(Wk, wh + (size_t)DMOD * DMOD, w4);
        cvt_f32_to_f16_x4<<<w4 / 256, 256, 0, stream>>>(Wv, wh + (size_t)2 * DMOD * DMOD, w4);
    }

    // sigma (exact fp32)
    sigma_kernel<<<NTOK * 32 / 256, 256, 0, stream>>>(x, Ws, sigma);

    // zero the row-sum accumulator (graph-capturable memset node)
    (void)hipMemsetAsync(rowsum, 0, (size_t)NTOK * 4, stream);

    // K1: Q,K (row-major f16) and V (transposed f16)
    qkv_kernel<<<dim3(NTOK / 16, DMOD / 64, 3), 32, 0, stream>>>(xh, wh, qh, kmh, vt);

    // K2a: E = exp(QK^T/sqrt(d)) into S region + row sums (LDS-staged B)
    scores_kernel<<<dim3(NTOK / 64, NTOK / 64), 128, 0, stream>>>(qh, kmh, S, rowsum);

    // K2b: normalize S in place
    snorm_kernel<<<(size_t)NTOK * NTOK / 4 / 256, 256, 0, stream>>>(S, rowsum);

    // K3: Z = S @ V (LDS-staged Vt)
    zout_kernel<<<dim3(NTOK / 64, DMOD / 64), 128, 0, stream>>>(S, vt, Z);

    // K4: prior association P
    prior_kernel<<<NTOK, 256, 0, stream>>>(sigma, P);

    (void)in_sizes; (void)n_in; (void)out_size; (void)ws_size;
}
